// SparsifinerLVViT_81011673137615
// MI455X (gfx1250) — matrix operand order repaired and checked by reading, not compiled, not run on hardware
//
#include <hip/hip_runtime.h>
#include <cstdint>
#include <cstddef>

// ---------------- problem constants ----------------
#define B_    64
#define N_    197
#define C_    768
#define H_    12
#define CH_   64
#define RC_   32
#define RN_   24
#define NPAD  224          // keys/queries padded to 14*16 for WMMA tiling
#define BN_   (B_*N_)      // 12608 rows (= 788 * 16 exactly)
#define BH_   (B_*H_)      // 768 head-batches
#define FC1_  3072
#define BUDGET_ 50
#define TOPK_   8
#define MP_SCALE_   0.28867513459481287f   // 12^-0.5
#define ATTN_SCALE_ 0.125f                 // 64^-0.5
#define BASIS_TH_   0.02f
#define LN_EPS_     1e-5f

typedef __bf16 bf16_t;
typedef __attribute__((ext_vector_type(16))) __bf16 v16bf;
typedef __attribute__((ext_vector_type(8)))  __bf16 v8bf;
typedef __attribute__((ext_vector_type(8)))  float  v8f;

// ---------------- WMMA fragment helpers (gfx1250 wave32 layouts) ----------------
// A (16x32 bf16, MxK): lanes 0-15 row M=lane, halves 0-7 = K 0..7, halves 8-15 = K 16..23
//                      lanes 16-31 row M=lane-16, halves = K 8..15 / 24..31
__device__ inline v16bf load_a_frag(const bf16_t* A, int lda, int m0, int k0, int lane) {
    const int row = m0 + (lane & 15);
    const int sel = (lane >> 4) * 8;
    const bf16_t* p = A + (size_t)row * lda + k0 + sel;
    union { v16bf v; v8bf h[2]; } f;
    f.h[0] = *(const v8bf*)(p);
    f.h[1] = *(const v8bf*)(p + 16);
    return f.v;
}
__device__ inline v16bf load_a_frag_clamp(const bf16_t* A, int lda, int m0, int k0, int lane,
                                          int maxRow) {
    int row = m0 + (lane & 15);
    row = row < maxRow ? row : maxRow;
    const int sel = (lane >> 4) * 8;
    const bf16_t* p = A + (size_t)row * lda + k0 + sel;
    union { v16bf v; v8bf h[2]; } f;
    f.h[0] = *(const v8bf*)(p);
    f.h[1] = *(const v8bf*)(p + 16);
    return f.v;
}
// B (32x16 bf16, KxN) sourced from a row-major "B-transposed" array (rows = output cols):
// lane L: column N = n0+(L&15), K chunks k0 + (L>>4)*8 .. +7 and k0 + 16 + (L>>4)*8 .. +7
__device__ inline v16bf load_b_frag(const bf16_t* BT, int ldb, int n0, int k0, int lane) {
    const bf16_t* p = BT + (size_t)(n0 + (lane & 15)) * ldb + k0 + (lane >> 4) * 8;
    union { v16bf v; v8bf h[2]; } f;
    f.h[0] = *(const v8bf*)(p);
    f.h[1] = *(const v8bf*)(p + 16);
    return f.v;
}
__device__ inline v8f wmma_bf16(v16bf a, v16bf b, v8f c) {
    return __builtin_amdgcn_wmma_f32_16x16x32_bf16(false, a, false, b, (short)0, c, false, false);
}

// ---------------- utility kernels ----------------
__global__ void zero_b16(bf16_t* p, size_t n) {
    size_t i = (size_t)blockIdx.x * 256 + threadIdx.x;
    if (i < n) p[i] = (bf16_t)0.f;
}
__global__ void zero_u32(unsigned* p, size_t n) {
    size_t i = (size_t)blockIdx.x * 256 + threadIdx.x;
    if (i < n) p[i] = 0u;
}
// W (K x Nc f32, row-major) -> WT (Nc x K bf16, row-major)
__global__ void wconv_kernel(const float* __restrict__ W, bf16_t* __restrict__ WT, int K, int Nc) {
    size_t idx = (size_t)blockIdx.x * 256 + threadIdx.x;
    size_t total = (size_t)K * Nc;
    if (idx < total) {
        int kq = (int)(idx / Nc), nc = (int)(idx % Nc);
        WT[(size_t)nc * K + kq] = (bf16_t)W[idx];
    }
}

// ---------------- LayerNorm (f32 in -> bf16 out), one wave per 768-wide row ----------------
__global__ __launch_bounds__(256)
void ln_kernel(const float* __restrict__ x, const float* __restrict__ g,
               const float* __restrict__ b, bf16_t* __restrict__ out, int rows) {
    const int wave = threadIdx.x >> 5, lane = threadIdx.x & 31;
    const int row = blockIdx.x * 8 + wave;
    if (row >= rows) return;
    const float* xr = x + (size_t)row * C_;
    float vals[24]; float s = 0.f;
    #pragma unroll
    for (int i = 0; i < 24; i++) { vals[i] = xr[i * 32 + lane]; s += vals[i]; }
    #pragma unroll
    for (int off = 16; off; off >>= 1) s += __shfl_xor(s, off, 32);
    const float mu = s * (1.f / 768.f);
    float v = 0.f;
    #pragma unroll
    for (int i = 0; i < 24; i++) { const float d = vals[i] - mu; v += d * d; }
    #pragma unroll
    for (int off = 16; off; off >>= 1) v += __shfl_xor(v, off, 32);
    const float rs = rsqrtf(v * (1.f / 768.f) + LN_EPS_);
    bf16_t* orow = out + (size_t)row * C_;
    #pragma unroll
    for (int i = 0; i < 24; i++) {
        const int c = i * 32 + lane;
        orow[c] = (bf16_t)((vals[i] - mu) * rs * g[c] + b[c]);
    }
}

// ---------------- generic WMMA GEMM ------------------------------------------------------------
// Each wave owns a 32x64 output tile (2 A-frags x 4 B-frags = 8 WMMAs per K-step; every B
// fragment is reused by both row tiles -> ~22 FLOP/byte of fragment traffic).
// Block = 4 waves = 128 rows; grid.x = ceil(788/8) = 99 row-blocks (stores guarded, loads clamped).
// modes: 0 = QKV scatter to q/k/vT, 1 = +bias+resid -> f32, 2 = +bias + GELU -> bf16 (ld 3072),
//        3 = +bias+resid -> f32 (final output)
__global__ __launch_bounds__(128)
void gemm_wmma(const bf16_t* __restrict__ A, const bf16_t* __restrict__ BT, int K, int mode,
               const float* __restrict__ bias, const float* __restrict__ resid,
               float* __restrict__ outf, bf16_t* __restrict__ outb,
               bf16_t* __restrict__ qd, bf16_t* __restrict__ kd, bf16_t* __restrict__ vtd) {
    const int wave = threadIdx.x >> 5, lane = threadIdx.x & 31;
    const int m0 = (blockIdx.x * 8 + wave * 2) * 16;     // two 16-row tiles per wave
    const int n0 = blockIdx.y * 64;
    const v8f zero = {0.f, 0.f, 0.f, 0.f, 0.f, 0.f, 0.f, 0.f};
    v8f acc[2][4] = {{zero, zero, zero, zero}, {zero, zero, zero, zero}};
    for (int k0 = 0; k0 < K; k0 += 32) {
        const v16bf a0 = load_a_frag_clamp(A, K, m0,      k0, lane, BN_ - 1);
        const v16bf a1 = load_a_frag_clamp(A, K, m0 + 16, k0, lane, BN_ - 1);
        #pragma unroll
        for (int t = 0; t < 4; t++) {
            const v16bf b = load_b_frag(BT, K, n0 + t * 16, k0, lane);
            acc[0][t] = wmma_bf16(a0, b, acc[0][t]);
            acc[1][t] = wmma_bf16(a1, b, acc[1][t]);
        }
    }
    const int half = lane >> 4, lcol = lane & 15;
    #pragma unroll
    for (int u = 0; u < 2; u++) {
        #pragma unroll
        for (int t = 0; t < 4; t++) {
            const int n = n0 + t * 16 + lcol;
            #pragma unroll
            for (int r = 0; r < 8; r++) {
                const int m = m0 + u * 16 + r + 8 * half;
                if (m >= BN_) continue;
                const float val = acc[u][t][r];
                if (mode == 0) {                      // QKV epilogue, Nc = 2304
                    const int which = n / C_;
                    const int rem = n - which * C_;
                    const int head = rem >> 6, ch = rem & 63;
                    const int bb = m / N_, nn = m - bb * N_;
                    const size_t bh = (size_t)bb * H_ + head;
                    if (which == 0)      qd[(bh * NPAD + nn) * CH_ + ch] = (bf16_t)val;
                    else if (which == 1) kd[(bh * NPAD + nn) * CH_ + ch] = (bf16_t)val;
                    else                 vtd[(bh * CH_ + ch) * NPAD + nn] = (bf16_t)val;
                } else if (mode == 2) {               // FC1 + GELU(exact) -> bf16
                    const float xg = val + bias[n];
                    outb[(size_t)m * FC1_ + n] =
                        (bf16_t)(0.5f * xg * (1.f + erff(xg * 0.70710678118654752f)));
                } else {                              // proj / FC2: + bias + residual -> f32
                    const size_t idx = (size_t)m * C_ + n;
                    outf[idx] = val + bias[n] + resid[idx];
                }
            }
        }
    }
}

// ---------------- qp/kp head projections on WMMA -----------------------------------------------
// One wave per (bh, 16-row tile, which-tensor): D(16x32) = src(16x64) @ WT(64x32), K = 64.
__global__ __launch_bounds__(32)
void qpkp_wmma(const bf16_t* __restrict__ q, const bf16_t* __restrict__ k,
               const bf16_t* __restrict__ WcqT, const bf16_t* __restrict__ WckT,
               const float* __restrict__ bcq, const float* __restrict__ bck,
               float* __restrict__ qp, float* __restrict__ kp) {
    const int lane = threadIdx.x & 31;
    const int m0 = blockIdx.x * 16;                  // row tile inside the padded 224-row block
    const int bh = blockIdx.y;
    const int which = blockIdx.z;
    const bf16_t* src = (which ? k : q) + (size_t)bh * NPAD * CH_;
    const bf16_t* WT = which ? WckT : WcqT;
    const float* bias = which ? bck : bcq;
    float* dst = which ? kp : qp;

    const v8f zero = {0.f, 0.f, 0.f, 0.f, 0.f, 0.f, 0.f, 0.f};
    v8f acc[2] = {zero, zero};
    #pragma unroll
    for (int k0 = 0; k0 < CH_; k0 += 32) {
        const v16bf a = load_a_frag(src, CH_, m0, k0, lane);
        #pragma unroll
        for (int t = 0; t < 2; t++)
            acc[t] = wmma_bf16(a, load_b_frag(WT, CH_, t * 16, k0, lane), acc[t]);
    }
    const int half = lane >> 4, lcol = lane & 15;
    #pragma unroll
    for (int t = 0; t < 2; t++) {
        const int c = t * 16 + lcol;
        #pragma unroll
        for (int r = 0; r < 8; r++) {
            const int nn = m0 + r + 8 * half;
            if (nn < N_)
                dst[((size_t)bh * N_ + nn) * RC_ + c] = acc[t][r] + bias[c];
        }
    }
}

// ---------------- kk[bh,c,r] = sum_n kp[bh,n,c] * proj_n[n,r] ----------------
__global__ __launch_bounds__(768)
void kk_kernel(const float* __restrict__ kp, const float* __restrict__ proj_n,
               float* __restrict__ kkb) {
    const int bh = blockIdx.x, t = threadIdx.x;
    if (t >= RC_ * RN_) return;
    const int c = t / RN_, r = t % RN_;
    const float* kpb = kp + (size_t)bh * N_ * RC_;
    float s = 0.f;
    for (int n = 0; n < N_; n++) s += kpb[(size_t)n * RC_ + c] * proj_n[(size_t)n * RN_ + r];
    kkb[(size_t)bh * RC_ * RN_ + t] = s;
}

// ---------------- fused: cheap attn -> softmax -> top-8 -> basis expand -> top-50 mask ----------
// one wave per query row; grid (ceil(197/4), BH_)
__global__ __launch_bounds__(128)
void mask_kernel(const float* __restrict__ qp, const float* __restrict__ kkb,
                 const float* __restrict__ proj_back, unsigned* __restrict__ maskb) {
    __shared__ float skk[RC_ * RN_];
    __shared__ float approx[4][NPAD];
    const int wave = threadIdx.x >> 5, lane = threadIdx.x & 31;
    const int bh = blockIdx.y;
    const int n = blockIdx.x * 4 + wave;

    for (int i = threadIdx.x; i < RC_ * RN_; i += 128) skk[i] = kkb[(size_t)bh * RC_ * RN_ + i];
    __syncthreads();

    const bool activeRow = (n >= 1) && (n < N_);
    float coef = 0.f;
    if (activeRow) {
        const float qv = qp[((size_t)bh * N_ + n) * RC_ + lane];   // lane == RC channel
        float ch = 0.f;
        #pragma unroll
        for (int c = 0; c < RC_; c++) {
            const float qc = __shfl(qv, c, 32);
            if (lane < RN_) ch += qc * skk[c * RN_ + lane];
        }
        ch *= MP_SCALE_;
        float mx = (lane < RN_) ? ch : -3.0e38f;
        #pragma unroll
        for (int off = 16; off; off >>= 1) mx = fmaxf(mx, __shfl_xor(mx, off, 32));
        const float e = (lane < RN_) ? __expf(ch - mx) : 0.f;
        float s = e;
        #pragma unroll
        for (int off = 16; off; off >>= 1) s += __shfl_xor(s, off, 32);
        coef = e / s;
        int cnt = 0;                                // keep iff fewer than 8 strictly greater
        #pragma unroll
        for (int r = 0; r < RN_; r++) cnt += (__shfl(coef, r, 32) > coef) ? 1 : 0;
        if (cnt >= TOPK_ || lane >= RN_) coef = 0.f;
    }
    #pragma unroll
    for (int it = 0; it < 7; it++) {                // approx_attn row, 197 entries
        const int j = it * 32 + lane;
        float a = 0.f;
        if (activeRow && j < N_) {
            #pragma unroll
            for (int r = 0; r < RN_; r++) {
                const float cr = __shfl(coef, r, 32);
                float bj = fabsf(proj_back[(size_t)j * RN_ + r]);
                bj = (bj > BASIS_TH_) ? bj : 0.f;
                a += cr * bj;
            }
        }
        approx[wave][it * 32 + lane] = a;
    }
    __syncthreads();

    unsigned* mrow = maskb + ((size_t)bh * NPAD + (n < N_ ? n : 0)) * 8;
    if (n == 0 && lane < 7)                         // CLS row: all keys kept
        mrow[lane] = (lane < 6) ? 0xFFFFFFFFu : 0x1Fu;
    #pragma unroll
    for (int it = 0; it < 7; it++) {                // top-50 via rank count + ballot
        const int j = it * 32 + lane;
        const float mine = approx[wave][it * 32 + lane];
        int cnt = 0;
        for (int t = 0; t < N_; t++) cnt += (approx[wave][t] > mine) ? 1 : 0;
        const bool keep = activeRow && (j < N_) && (cnt < BUDGET_);
        const unsigned word = (unsigned)__ballot(keep);
        if (activeRow && lane == 0) mrow[it] = word;
    }
}

// ---------------- flash attention: one wave per (head-batch, 16-query tile) ----------------
__global__ __launch_bounds__(32)
void attn_kernel(const bf16_t* __restrict__ q, const bf16_t* __restrict__ k,
                 const bf16_t* __restrict__ vT, const unsigned* __restrict__ maskb,
                 bf16_t* __restrict__ o) {
    __shared__ bf16_t pl[16 * 32];
    const int lane = threadIdx.x & 31;
    const int bh = blockIdx.y;
    const int m0 = blockIdx.x * 16;
    const int half = lane >> 4, lcol = lane & 15;

    const bf16_t* qb = q + (size_t)bh * NPAD * CH_;
    const bf16_t* kb = k + (size_t)bh * NPAD * CH_;
    const bf16_t* vb = vT + (size_t)bh * CH_ * NPAD;
    const unsigned* mb = maskb + (size_t)bh * NPAD * 8;

    const v16bf aq0 = load_a_frag(qb, CH_, m0, 0, lane);
    const v16bf aq1 = load_a_frag(qb, CH_, m0, 32, lane);

    const v8f zero = {0.f, 0.f, 0.f, 0.f, 0.f, 0.f, 0.f, 0.f};
    v8f acc[4] = {zero, zero, zero, zero};
    float rm[8], rl[8];
    #pragma unroll
    for (int r = 0; r < 8; r++) { rm[r] = -3.0e38f; rl[r] = 0.f; }

    for (int kk0 = 0; kk0 < NPAD; kk0 += 32) {
        v8f s0 = zero, s1 = zero;                               // 16q x 32keys scores
        s0 = wmma_bf16(aq0, load_b_frag(kb, CH_, kk0,      0,  lane), s0);
        s0 = wmma_bf16(aq1, load_b_frag(kb, CH_, kk0,      32, lane), s0);
        s1 = wmma_bf16(aq0, load_b_frag(kb, CH_, kk0 + 16, 0,  lane), s1);
        s1 = wmma_bf16(aq1, load_b_frag(kb, CH_, kk0 + 16, 32, lane), s1);

        const int key0 = kk0 + lcol, key1 = kk0 + 16 + lcol;
        float sv0[8], sv1[8], tmax[8];
        #pragma unroll
        for (int r = 0; r < 8; r++) {
            const int m = m0 + r + 8 * half;
            const unsigned w0 = mb[(size_t)m * 8 + (key0 >> 5)];
            const unsigned w1 = mb[(size_t)m * 8 + (key1 >> 5)];
            sv0[r] = ((w0 >> (key0 & 31)) & 1u) ? s0[r] * ATTN_SCALE_ : -1.0e30f;
            sv1[r] = ((w1 >> (key1 & 31)) & 1u) ? s1[r] * ATTN_SCALE_ : -1.0e30f;
            tmax[r] = fmaxf(sv0[r], sv1[r]);
        }
        #pragma unroll
        for (int off = 1; off < 16; off <<= 1)
            #pragma unroll
            for (int r = 0; r < 8; r++)
                tmax[r] = fmaxf(tmax[r], __shfl_xor(tmax[r], off, 32));

        float p0[8], p1[8];
        #pragma unroll
        for (int r = 0; r < 8; r++) {                           // online softmax update
            const float nm = fmaxf(rm[r], tmax[r]);
            const float sc = __expf(rm[r] - nm);
            rm[r] = nm;
            rl[r] *= sc;
            #pragma unroll
            for (int t = 0; t < 4; t++) acc[t][r] *= sc;
            p0[r] = __expf(sv0[r] - nm);
            p1[r] = __expf(sv1[r] - nm);
            float ps = p0[r] + p1[r];
            #pragma unroll
            for (int off = 1; off < 16; off <<= 1) ps += __shfl_xor(ps, off, 32);
            rl[r] += ps;
        }

        __syncthreads();                                        // prior A-frag reads done
        #pragma unroll
        for (int r = 0; r < 8; r++) {                           // P: C-layout -> LDS (row-major 16x32)
            const int row = r + 8 * half;
            pl[row * 32 + lcol]      = (bf16_t)p0[r];
            pl[row * 32 + 16 + lcol] = (bf16_t)p1[r];
        }
        __syncthreads();
        union { v16bf v; v8bf h[2]; } pf;                       // reload as A fragment
        const bf16_t* pp = &pl[lcol * 32 + half * 8];
        pf.h[0] = *(const v8bf*)pp;
        pf.h[1] = *(const v8bf*)(pp + 16);
        #pragma unroll
        for (int t = 0; t < 4; t++)                             // O += P @ V (V transposed layout)
            acc[t] = wmma_bf16(pf.v, load_b_frag(vb, NPAD, t * 16, kk0, lane), acc[t]);
    }

    const int bb = bh / H_, hh = bh % H_;
    #pragma unroll
    for (int r = 0; r < 8; r++) {
        const int n = m0 + r + 8 * half;
        if (n < N_) {
            const float inv = 1.f / rl[r];
            #pragma unroll
            for (int t = 0; t < 4; t++)
                o[(size_t)(bb * N_ + n) * C_ + hh * CH_ + t * 16 + lcol] = (bf16_t)(acc[t][r] * inv);
        }
    }
}

// ---------------- host launcher ----------------
extern "C" void kernel_launch(void* const* d_in, const int* in_sizes, int n_in,
                              void* d_out, int out_size, void* d_ws, size_t ws_size,
                              hipStream_t stream) {
    const float* x        = (const float*)d_in[0];
    const float* Wqkv     = (const float*)d_in[1];
    const float* Wcq      = (const float*)d_in[2];
    const float* bcq      = (const float*)d_in[3];
    const float* Wck      = (const float*)d_in[4];
    const float* bck      = (const float*)d_in[5];
    const float* proj_n   = (const float*)d_in[6];
    const float* proj_bk  = (const float*)d_in[7];
    const float* Wproj    = (const float*)d_in[8];
    const float* bproj    = (const float*)d_in[9];
    const float* g1       = (const float*)d_in[10];
    const float* b1       = (const float*)d_in[11];
    const float* g2       = (const float*)d_in[12];
    const float* b2       = (const float*)d_in[13];
    const float* W1       = (const float*)d_in[14];
    const float* bfc1     = (const float*)d_in[15];
    const float* W2       = (const float*)d_in[16];
    const float* bfc2     = (const float*)d_in[17];
    float* outp = (float*)d_out;

    char* base = (char*)d_ws;
    size_t off = 0;
    auto take = [&](size_t bytes) -> char* {
        char* p = base + off;
        off += (bytes + 255) & ~(size_t)255;
        return p;
    };
    bf16_t*   hbf    = (bf16_t*)take((size_t)BN_ * C_ * 2);            // LN1 output
    bf16_t*   WqkvT  = (bf16_t*)take((size_t)3 * C_ * C_ * 2);         // 2304 x 768
    bf16_t*   WprojT = (bf16_t*)take((size_t)C_ * C_ * 2);
    bf16_t*   W1T    = (bf16_t*)take((size_t)FC1_ * C_ * 2);           // 3072 x 768
    bf16_t*   W2T    = (bf16_t*)take((size_t)C_ * FC1_ * 2);           // 768 x 3072
    bf16_t*   WcqT   = (bf16_t*)take((size_t)RC_ * CH_ * 2);           // 32 x 64
    bf16_t*   WckT   = (bf16_t*)take((size_t)RC_ * CH_ * 2);
    bf16_t*   qd     = (bf16_t*)take((size_t)BH_ * NPAD * CH_ * 2);
    bf16_t*   kd     = (bf16_t*)take((size_t)BH_ * NPAD * CH_ * 2);
    bf16_t*   vtd    = (bf16_t*)take((size_t)BH_ * CH_ * NPAD * 2);
    float*    qp     = (float*)take((size_t)BH_ * N_ * RC_ * 4);
    float*    kp     = (float*)take((size_t)BH_ * N_ * RC_ * 4);
    float*    kkb    = (float*)take((size_t)BH_ * RC_ * RN_ * 4);
    unsigned* maskb  = (unsigned*)take((size_t)BH_ * NPAD * 8 * 4);
    bf16_t*   ob     = (bf16_t*)take((size_t)BN_ * C_ * 2);            // attention out (B,N,C)
    float*    x1     = (float*)take((size_t)BN_ * C_ * 4);             // residual 1
    bf16_t*   h2     = (bf16_t*)take((size_t)BN_ * C_ * 2);            // LN2 output
    bf16_t*   mid    = (bf16_t*)take((size_t)BN_ * FC1_ * 2);          // GELU(fc1)

    auto blocks = [](size_t n, int t) { return (unsigned)((n + (size_t)t - 1) / t); };
    const unsigned GX = (788 + 7) / 8;   // 99 row-blocks of 128 rows (guarded)

    // 0) zero padded q/k/vT and mask (padding keys must read as 0 / masked)
    const size_t qkvN = (size_t)BH_ * NPAD * CH_;
    zero_b16<<<blocks(qkvN, 256), 256, 0, stream>>>(qd, qkvN);
    zero_b16<<<blocks(qkvN, 256), 256, 0, stream>>>(kd, qkvN);
    zero_b16<<<blocks(qkvN, 256), 256, 0, stream>>>(vtd, qkvN);
    const size_t maskN = (size_t)BH_ * NPAD * 8;
    zero_u32<<<blocks(maskN, 256), 256, 0, stream>>>(maskb, maskN);

    // 1) weight convert + transpose to bf16
    wconv_kernel<<<blocks((size_t)C_ * 3 * C_, 256), 256, 0, stream>>>(Wqkv, WqkvT, C_, 3 * C_);
    wconv_kernel<<<blocks((size_t)C_ * C_, 256), 256, 0, stream>>>(Wproj, WprojT, C_, C_);
    wconv_kernel<<<blocks((size_t)C_ * FC1_, 256), 256, 0, stream>>>(W1, W1T, C_, FC1_);
    wconv_kernel<<<blocks((size_t)FC1_ * C_, 256), 256, 0, stream>>>(W2, W2T, FC1_, C_);
    wconv_kernel<<<blocks((size_t)CH_ * RC_, 256), 256, 0, stream>>>(Wcq, WcqT, CH_, RC_);
    wconv_kernel<<<blocks((size_t)CH_ * RC_, 256), 256, 0, stream>>>(Wck, WckT, CH_, RC_);

    // 2) LN1
    ln_kernel<<<BN_ / 8, 256, 0, stream>>>(x, g1, b1, hbf, BN_);

    // 3) QKV GEMM (WMMA) -> q, k, vT (bf16, padded layouts)
    gemm_wmma<<<dim3(GX, (3 * C_) / 64), 128, 0, stream>>>(hbf, WqkvT, C_, 0,
        nullptr, nullptr, nullptr, nullptr, qd, kd, vtd);

    // 4) cheap-path projections (WMMA) and rank mixing
    qpkp_wmma<<<dim3(NPAD / 16, BH_, 2), 32, 0, stream>>>(qd, kd, WcqT, WckT, bcq, bck, qp, kp);
    kk_kernel<<<BH_, RC_ * RN_, 0, stream>>>(kp, proj_n, kkb);

    // 5) fused sparsity-mask builder (softmax/top-8/basis/top-50 budget)
    mask_kernel<<<dim3((N_ + 3) / 4, BH_), 128, 0, stream>>>(qp, kkb, proj_bk, maskb);

    // 6) masked flash attention (WMMA) -> ob (B,N,C) bf16
    attn_kernel<<<dim3(13, BH_), 32, 0, stream>>>(qd, kd, vtd, maskb, ob);

    // 7) output projection + bias + residual -> x1 (f32)
    gemm_wmma<<<dim3(GX, C_ / 64), 128, 0, stream>>>(ob, WprojT, C_, 1,
        bproj, x, x1, nullptr, nullptr, nullptr, nullptr);

    // 8) LN2
    ln_kernel<<<BN_ / 8, 256, 0, stream>>>(x1, g2, b2, h2, BN_);

    // 9) FC1 + GELU -> mid (bf16)
    gemm_wmma<<<dim3(GX, FC1_ / 64), 128, 0, stream>>>(h2, W1T, C_, 2,
        bfc1, nullptr, nullptr, mid, nullptr, nullptr, nullptr);

    // 10) FC2 + bias + residual -> final output (f32)
    gemm_wmma<<<dim3(GX, C_ / 64), 128, 0, stream>>>(mid, W2T, FC1_, 3,
        bfc2, x1, outp, nullptr, nullptr, nullptr, nullptr);

    (void)in_sizes; (void)n_in; (void)out_size; (void)ws_size;
}
